// network_template_62294205662141
// MI455X (gfx1250) — compile-verified, matrix operand order checked by reading, not run on hardware
//
#include <hip/hip_runtime.h>
#include <hip/hip_bf16.h>

// ---------------------------------------------------------------------------
// Problem constants (match the reference)
// ---------------------------------------------------------------------------
#define NN      20000
#define EE      160000
#define DIM_IN  6
#define LDIM    128
#define ADIM    8
#define NLAYERS 4
#define EIN     273           // 2*L + 2*A + 1
#define EINP    288           // padded to 9 * 32
#define KNODE   256           // 2*L
#define MAX_R   50.0f

#define WPB     8             // waves per block (256 threads): 2 waves / SIMD32

// per-layer transposed-f16 weight blob sizes (in halfs)
#define EDGE_BLOB_H (128 * EINP + 128 * 128)    // 53248 halfs = 106496 B
#define NODE_BLOB_H (128 * KNODE + 128 * 128)   // 49152 halfs =  98304 B

typedef _Float16 v16h __attribute__((ext_vector_type(16)));
typedef _Float16 v8h  __attribute__((ext_vector_type(8)));
typedef _Float16 v4h  __attribute__((ext_vector_type(4)));
typedef float    v8f  __attribute__((ext_vector_type(8)));

typedef unsigned int u32x4 __attribute__((ext_vector_type(4)));
typedef int          i32x4 __attribute__((ext_vector_type(4)));
typedef int          i32x8 __attribute__((ext_vector_type(8)));

#if defined(__has_builtin)
# if __has_builtin(__builtin_amdgcn_tensor_load_to_lds)
#  define HAVE_TDM 1
# endif
#endif
#ifndef HAVE_TDM
# define HAVE_TDM 0
#endif

// ---------------------------------------------------------------------------
// TDM: 1-D contiguous tensor DMA, global -> LDS.  Descriptor per
// CDNA5 ISA: group0 = {count|flags, lds_addr,
// global_addr[56:0], type=2}; group1 packs data_size=4B, tensor_dim0,
// tile_dim0 (= dwords, < 65536), tile_dim1 = tensor_dim1 = 1.
// Issued by one wave (EXEC ignored; per-wave op), tracked by TENSORcnt.
// ---------------------------------------------------------------------------
#if HAVE_TDM
__device__ __forceinline__ void tdm_load_1d(const void* gsrc, void* lds_dst,
                                            unsigned bytes) {
  const unsigned n4 = bytes >> 2;                       // dwords
  const unsigned long long ga = (unsigned long long)(uintptr_t)gsrc;
  const unsigned lds = (unsigned)(uintptr_t)lds_dst;    // low 32 bits = LDS offset
  u32x4 g0;
  g0[0] = 1u;                                           // count=1 (valid), user mode
  g0[1] = lds;                                          // lds_addr
  g0[2] = (unsigned)ga;                                 // global_addr[31:0]
  g0[3] = (unsigned)((ga >> 32) & 0x1FFFFFFu) | (2u << 30); // addr[56:32]|type=2
  i32x8 g1;
  g1[0] = (int)(2u << 16);                              // data_size=2 (4 bytes)
  g1[1] = (int)((n4 & 0xFFFFu) << 16);                  // tensor_dim0 lo16
  g1[2] = (int)((n4 >> 16) & 0xFFFFu) | (1 << 16);      // tensor_dim0 hi16 | dim1 lo16 = 1
  g1[3] = (int)((n4 & 0xFFFFu) << 16);                  // tile_dim0
  g1[4] = 1;                                            // tile_dim1 = 1
  g1[5] = (int)n4;                                      // tensor_dim0_stride lo32
  g1[6] = 0;
  g1[7] = 0;
  i32x4 z4 = {0, 0, 0, 0};
#if __clang_major__ >= 23
  i32x8 z8 = {0, 0, 0, 0, 0, 0, 0, 0};
  __builtin_amdgcn_tensor_load_to_lds(g0, g1, z4, z4, z8, 0);
#else
  __builtin_amdgcn_tensor_load_to_lds(g0, g1, z4, z4, 0);
#endif
}
#endif

__device__ __forceinline__ void tdm_wait0() {
#if defined(__has_builtin) && __has_builtin(__builtin_amdgcn_s_wait_tensorcnt)
  __builtin_amdgcn_s_wait_tensorcnt(0);
#else
  asm volatile("s_wait_tensorcnt 0" ::: "memory");
#endif
}

// stage a contiguous per-layer weight blob (bytes % 16 == 0) into LDS
__device__ __forceinline__ void stage_weights(const _Float16* __restrict__ gw,
                                              _Float16* __restrict__ smem,
                                              unsigned bytes) {
#if HAVE_TDM
  if (threadIdx.x < 32) {       // wave 0 issues the DMA and waits TENSORcnt
    tdm_load_1d(gw, smem, bytes);
    tdm_wait0();
  }
  __syncthreads();
#else
  const uint4* s = (const uint4*)gw;
  uint4*       d = (uint4*)smem;
  for (unsigned idx = threadIdx.x; idx < (bytes >> 4); idx += blockDim.x)
    d[idx] = s[idx];
  __syncthreads();
#endif
}

// ---------------------------------------------------------------------------
// WMMA fragment loader (CDNA5 ISA, 16-bit A-matrix 16x32 layout):
//   lanes 0-15 : v[0..7] = K  0..7 , v[8..15] = K 16..23
//   lanes 16-31: v[0..7] = K  8..15, v[8..15] = K 24..31
// Weights stored transposed [N][K] so the same loader serves A and B.
// ---------------------------------------------------------------------------
__device__ __forceinline__ v16h load_frag(const _Float16* __restrict__ row,
                                          int k0, int half) {
  const v8h lo = *(const v8h*)(row + k0 + half * 8);
  const v8h hi = *(const v8h*)(row + k0 + 16 + half * 8);
  v16h a;
#pragma unroll
  for (int i = 0; i < 8; ++i) { a[i] = lo[i]; a[i + 8] = hi[i]; }
  return a;
}

__device__ __forceinline__ v8f wmma_f16(v16h a, v16h b, v8f c) {
  return __builtin_amdgcn_wmma_f32_16x16x32_f16(false, a, false, b,
                                                (short)0, c, false, false);
}

__device__ __forceinline__ float silu(float x) {
  return x * (1.0f / (1.0f + __expf(-x)));
}

// ---------------------------------------------------------------------------
// k_prep: transpose + f16-convert one weight matrix [K][128] -> [128][Kpad]
// (zero-padded K tail).  Runs once per launch per matrix; output blobs are
// contiguous so in-kernel staging is a single TDM descriptor.
// ---------------------------------------------------------------------------
__global__ void k_prep(const float* __restrict__ src, _Float16* __restrict__ dst,
                       int K, int Kpad) {
  int idx = blockIdx.x * blockDim.x + threadIdx.x;
  if (idx >= 128 * Kpad) return;
  int n = idx / Kpad;
  int k = idx - n * Kpad;
  dst[idx] = (k < K) ? (_Float16)src[k * LDIM + n] : (_Float16)0.0f;
}

// ---------------------------------------------------------------------------
// k_init: attr embedding gather, y0 = x @ W^T, y_old = y0
// ---------------------------------------------------------------------------
__global__ void k_init(const float* __restrict__ x,
                       const int* __restrict__ node_attr,
                       const float* __restrict__ embed,
                       const float* __restrict__ W,
                       float* __restrict__ y_cur,
                       float* __restrict__ y_prev,
                       _Float16* __restrict__ attr_h) {
  int tid = blockIdx.x * blockDim.x + threadIdx.x;
  if (tid >= NN * LDIM) return;
  int n = tid >> 7;
  int l = tid & 127;
  float acc = 0.f;
#pragma unroll
  for (int d = 0; d < DIM_IN; ++d)
    acc += x[n * DIM_IN + d] * W[l * DIM_IN + d];
  y_cur[tid]  = acc;
  y_prev[tid] = acc;
  if (l < ADIM) {
    int a = node_attr[n];
    attr_h[n * ADIM + l] = (_Float16)embed[a * ADIM + l];
  }
}

// k_half: refresh the f16 shadow of y (coalesced, once per layer)
__global__ void k_half(const float* __restrict__ y, _Float16* __restrict__ yh) {
  int idx = blockIdx.x * blockDim.x + threadIdx.x;
  if (idx >= NN * LDIM / 4) return;
  float4 v = ((const float4*)y)[idx];
  ((v4h*)yh)[idx] = (v4h){(_Float16)v.x, (_Float16)v.y,
                          (_Float16)v.z, (_Float16)v.w};
}

__global__ void k_zero(float* __restrict__ p, int count) {
  int tid = blockIdx.x * blockDim.x + threadIdx.x;
  if (tid < count) p[tid] = 0.f;
}

// ---------------------------------------------------------------------------
// k_edge: per-edge message MLP + scatter-add via L2-resident f32 atomics.
// One wave per 16-edge tile.  Dynamic LDS (halfs):
//   [0)      w1T|w2T blob : 53248   (TDM-staged, transposed [n][k])
//   [53248)  mi   : 8 waves x 16 x 288
//   [90112)  ms   : 8 waves x 16 x 128
// total 106496 halfs = 212992 B  (CDNA5: 320 KB LDS / WGP)
// ---------------------------------------------------------------------------
__global__ void __launch_bounds__(256, 1)
k_edge(const _Float16* __restrict__ yh,        // [N][128] f16 shadow of y
       const float* __restrict__ xpos,         // [N][6], cols 0..2 used
       const _Float16* __restrict__ attr_h,    // [N][8]
       const int* __restrict__ esrc,
       const int* __restrict__ edst,
       const _Float16* __restrict__ gw,        // per-layer blob: w1T | w2T
       const float* __restrict__ b1,
       const float* __restrict__ b2,
       float* __restrict__ agg,                // [N][128]
       int ntiles) {
  extern __shared__ _Float16 smem[];
  _Float16* w1T = smem;                        // 128*288
  _Float16* w2T = smem + 128 * EINP;           // 128*128
  const int wave = threadIdx.x >> 5;
  const int lane = threadIdx.x & 31;
  _Float16* mi = smem + EDGE_BLOB_H + wave * 16 * EINP;
  _Float16* ms = smem + EDGE_BLOB_H + WPB * 16 * EINP + wave * 16 * LDIM;

  stage_weights(gw, smem, EDGE_BLOB_H * 2);

  const int half  = lane >> 4;
  const int nwave = gridDim.x * WPB;
  const int wid   = blockIdx.x * WPB + wave;

  for (int tile = wid; tile < ntiles; tile += nwave) {
    const int e0 = tile * 16;

    // ---- gather m_in = [y_src | y_dst | attr_src | attr_dst | w | pad] ----
    for (int r = 0; r < 16; ++r) {
      const int e   = e0 + r;
      const int src = esrc[e];
      const int dst = edst[e];
      {
        // lanes 0-15 copy yh[src] (8 halfs each), lanes 16-31 copy yh[dst]
        const int sel = (lane < 16) ? src : dst;
        const int off = (lane & 15) * 8;
        v8h v = *(const v8h*)(yh + sel * LDIM + off);
        *(v8h*)(mi + r * EINP + ((lane < 16) ? 0 : LDIM) + off) = v;
      }
      if (lane < 8) {
        mi[r * EINP + 2 * LDIM + lane] = attr_h[src * ADIM + lane];
      } else if (lane < 16) {
        mi[r * EINP + 2 * LDIM + 8 + (lane - 8)] = attr_h[dst * ADIM + (lane - 8)];
      } else if (lane == 16) {
        // edge weight: smooth_cutoff(|p_s - p_d| / 50) / |p_s - p_d|
        float dx = xpos[src * DIM_IN + 0] - xpos[dst * DIM_IN + 0];
        float dy = xpos[src * DIM_IN + 1] - xpos[dst * DIM_IN + 1];
        float dz = xpos[src * DIM_IN + 2] - xpos[dst * DIM_IN + 2];
        float rl = __builtin_sqrtf(dx * dx + dy * dy + dz * dz);
        float u  = 2.0f * (rl * (1.0f / MAX_R) - 1.0f);
        float cy = 0.5f * (1.0f - __cosf(3.14159265358979f * u));
        if (u > 0.0f)  cy = 0.0f;
        if (u < -1.0f) cy = 1.0f;
        mi[r * EINP + 272] = (_Float16)(cy / rl);
      } else {
        mi[r * EINP + 273 + (lane - 17)] = (_Float16)0.0f;   // K padding
      }
    }
    asm volatile("s_wait_dscnt 0" ::: "memory");
    __builtin_amdgcn_wave_barrier();

    // ---- GEMM1: 16x288 @ 288x128, bias + silu, restage to LDS ----
    v16h af[9];
#pragma unroll
    for (int kc = 0; kc < 9; ++kc)
      af[kc] = load_frag(mi + (lane & 15) * EINP, kc * 32, half);

#pragma unroll
    for (int ct = 0; ct < 8; ++ct) {
      const int nCol = ct * 16 + (lane & 15);
      v8f acc = {0.f, 0.f, 0.f, 0.f, 0.f, 0.f, 0.f, 0.f};
#pragma unroll
      for (int kc = 0; kc < 9; ++kc) {
        v16h bf = load_frag(w1T + nCol * EINP, kc * 32, half);
        acc = wmma_f16(af[kc], bf, acc);
      }
      const float b1v = b1[nCol];
#pragma unroll
      for (int r = 0; r < 8; ++r) {
        float hval = silu(acc[r] + b1v);
        ms[(r + 8 * half) * LDIM + nCol] = (_Float16)hval;
      }
    }
    asm volatile("s_wait_dscnt 0" ::: "memory");
    __builtin_amdgcn_wave_barrier();

    // ---- GEMM2: 16x128 @ 128x128, bias, atomic scatter to agg[dst] ----
    v16h a2[4];
#pragma unroll
    for (int kc = 0; kc < 4; ++kc)
      a2[kc] = load_frag(ms + (lane & 15) * LDIM, kc * 32, half);

    int dstv[8];
#pragma unroll
    for (int r = 0; r < 8; ++r) dstv[r] = edst[e0 + r + 8 * half];

#pragma unroll
    for (int ct = 0; ct < 8; ++ct) {
      const int nCol = ct * 16 + (lane & 15);
      v8f acc = {0.f, 0.f, 0.f, 0.f, 0.f, 0.f, 0.f, 0.f};
#pragma unroll
      for (int kc = 0; kc < 4; ++kc) {
        v16h bf = load_frag(w2T + nCol * 128, kc * 32, half);
        acc = wmma_f16(a2[kc], bf, acc);
      }
      const float b2v = b2[nCol];
#pragma unroll
      for (int r = 0; r < 8; ++r)
        atomicAdd(&agg[dstv[r] * LDIM + nCol], acc[r] + b2v);
    }
  }
}

// ---------------------------------------------------------------------------
// k_node: node MLP on u = [y | agg] + fused leapfrog update
//   y_next = 2*y - y_prev - h^2 * MLP(u)
// Dynamic LDS (halfs): blob 49152 | ust 8x16x256 | ms 8x16x128
// total 98304 halfs = 196608 B
// ---------------------------------------------------------------------------
__global__ void __launch_bounds__(256, 1)
k_node(const _Float16* __restrict__ yh,
       const float* __restrict__ y,
       const float* __restrict__ y_prev,
       const float* __restrict__ agg,
       const _Float16* __restrict__ gw,        // per-layer blob: nw1T | nw2T
       const float* __restrict__ nb1,
       const float* __restrict__ nb2,
       const float* __restrict__ hvec, int layer,
       float* __restrict__ y_next,
       int ntiles) {
  extern __shared__ _Float16 smem[];
  _Float16* w1T = smem;                        // 128*256
  _Float16* w2T = smem + 128 * KNODE;          // 128*128
  const int wave = threadIdx.x >> 5;
  const int lane = threadIdx.x & 31;
  _Float16* ust = smem + NODE_BLOB_H + wave * 16 * KNODE;
  _Float16* ms  = smem + NODE_BLOB_H + WPB * 16 * KNODE + wave * 16 * LDIM;

  stage_weights(gw, smem, NODE_BLOB_H * 2);

  const float hv = hvec[layer];
  const float h2 = hv * hv;
  const int half  = lane >> 4;
  const int nwave = gridDim.x * WPB;
  const int wid   = blockIdx.x * WPB + wave;

  for (int tile = wid; tile < ntiles; tile += nwave) {
    const int n0 = tile * 16;

    // ---- stage u = [y | agg] as f16 ----
    for (int r = 0; r < 16; ++r) {
      const int node = n0 + r;
      if (lane < 16) {
        const int off = lane * 8;
        v8h v = *(const v8h*)(yh + node * LDIM + off);
        *(v8h*)(ust + r * KNODE + off) = v;
      } else {
        const int c = (lane - 16) * 8;
        float4 a0 = *(const float4*)(agg + node * LDIM + c);
        float4 a1 = *(const float4*)(agg + node * LDIM + c + 4);
        *(v8h*)(ust + r * KNODE + LDIM + c) =
            (v8h){(_Float16)a0.x, (_Float16)a0.y, (_Float16)a0.z, (_Float16)a0.w,
                  (_Float16)a1.x, (_Float16)a1.y, (_Float16)a1.z, (_Float16)a1.w};
      }
    }
    asm volatile("s_wait_dscnt 0" ::: "memory");
    __builtin_amdgcn_wave_barrier();

    // ---- GEMM1: 16x256 @ 256x128, bias + silu ----
    v16h af[8];
#pragma unroll
    for (int kc = 0; kc < 8; ++kc)
      af[kc] = load_frag(ust + (lane & 15) * KNODE, kc * 32, half);

#pragma unroll
    for (int ct = 0; ct < 8; ++ct) {
      const int nCol = ct * 16 + (lane & 15);
      v8f acc = {0.f, 0.f, 0.f, 0.f, 0.f, 0.f, 0.f, 0.f};
#pragma unroll
      for (int kc = 0; kc < 8; ++kc) {
        v16h bf = load_frag(w1T + nCol * KNODE, kc * 32, half);
        acc = wmma_f16(af[kc], bf, acc);
      }
      const float b1v = nb1[nCol];
#pragma unroll
      for (int r = 0; r < 8; ++r) {
        float hval = silu(acc[r] + b1v);
        ms[(r + 8 * half) * LDIM + nCol] = (_Float16)hval;
      }
    }
    asm volatile("s_wait_dscnt 0" ::: "memory");
    __builtin_amdgcn_wave_barrier();

    // ---- GEMM2 + leapfrog update ----
    v16h a2[4];
#pragma unroll
    for (int kc = 0; kc < 4; ++kc)
      a2[kc] = load_frag(ms + (lane & 15) * LDIM, kc * 32, half);

#pragma unroll
    for (int ct = 0; ct < 8; ++ct) {
      const int nCol = ct * 16 + (lane & 15);
      v8f acc = {0.f, 0.f, 0.f, 0.f, 0.f, 0.f, 0.f, 0.f};
#pragma unroll
      for (int kc = 0; kc < 4; ++kc) {
        v16h bf = load_frag(w2T + nCol * 128, kc * 32, half);
        acc = wmma_f16(a2[kc], bf, acc);
      }
      const float b2v = nb2[nCol];
#pragma unroll
      for (int r = 0; r < 8; ++r) {
        const int node = n0 + r + 8 * half;
        float ynew = acc[r] + b2v;
        float ycur = y[node * LDIM + nCol];
        float yold = y_prev[node * LDIM + nCol];
        y_next[node * LDIM + nCol] = 2.0f * ycur - yold - h2 * ynew;
      }
    }
  }
}

// ---------------------------------------------------------------------------
// k_xupd: x = y @ W  (N x 128 @ 128 x 6) — negligible FLOPs, L2-resident
// ---------------------------------------------------------------------------
__global__ void k_xupd(const float* __restrict__ y,
                       const float* __restrict__ W,
                       float* __restrict__ xout) {
  int tid = blockIdx.x * blockDim.x + threadIdx.x;
  if (tid >= NN * DIM_IN) return;
  int n = tid / DIM_IN;
  int d = tid % DIM_IN;
  float acc = 0.f;
#pragma unroll 8
  for (int l = 0; l < LDIM; ++l)
    acc += y[n * LDIM + l] * W[l * DIM_IN + d];
  xout[tid] = acc;
}

__global__ void k_epi(const float* __restrict__ xb, float* __restrict__ out) {
  int tid = blockIdx.x * blockDim.x + threadIdx.x;
  if (tid < NN * DIM_IN) out[tid] = xb[tid];
  if (tid == 0) {
    out[NN * DIM_IN + 0] = -1.0f;
    out[NN * DIM_IN + 1] = -1.0f;
    out[NN * DIM_IN + 2] = 0.0f;
    out[NN * DIM_IN + 3] = 0.0f;
  }
}

// ---------------------------------------------------------------------------
// Host orchestration
// ---------------------------------------------------------------------------
extern "C" void kernel_launch(void* const* d_in, const int* in_sizes, int n_in,
                              void* d_out, int out_size, void* d_ws, size_t ws_size,
                              hipStream_t stream) {
  const float* x_in      = (const float*)d_in[0];
  const int*   node_attr = (const int*)d_in[2];
  const int*   esrc      = (const int*)d_in[3];
  const int*   edst      = (const int*)d_in[4];
  const float* W         = (const float*)d_in[5];
  const float* embed     = (const float*)d_in[6];
  const float* hvec      = (const float*)d_in[7];
  const float* msg_w1    = (const float*)d_in[8];
  const float* msg_b1    = (const float*)d_in[9];
  const float* msg_w2    = (const float*)d_in[10];
  const float* msg_b2    = (const float*)d_in[11];
  const float* node_w1   = (const float*)d_in[12];
  const float* node_b1   = (const float*)d_in[13];
  const float* node_w2   = (const float*)d_in[14];
  const float* node_b2   = (const float*)d_in[15];

  // ---- workspace carve-up (256-byte aligned regions) ----
  char* ws = (char*)d_ws;
  size_t off = 0;
  auto carve = [&](size_t bytes) {
    char* p = ws + off;
    off += (bytes + 255ull) & ~255ull;
    return p;
  };
  const size_t NY = (size_t)NN * LDIM * sizeof(float);
  float*    yA    = (float*)carve(NY);
  float*    yB    = (float*)carve(NY);
  float*    yC    = (float*)carve(NY);
  float*    agg   = (float*)carve(NY);
  float*    xb    = (float*)carve((size_t)NN * DIM_IN * sizeof(float));
  _Float16* attrh = (_Float16*)carve((size_t)NN * ADIM * 2);
  _Float16* yhb   = (_Float16*)carve((size_t)NN * LDIM * 2);
  _Float16* wedge = (_Float16*)carve((size_t)NLAYERS * EDGE_BLOB_H * 2);
  _Float16* wnode = (_Float16*)carve((size_t)NLAYERS * NODE_BLOB_H * 2);

  const int EDGE_LDS = (EDGE_BLOB_H + WPB * 16 * EINP  + WPB * 16 * LDIM) * 2;
  const int NODE_LDS = (NODE_BLOB_H + WPB * 16 * KNODE + WPB * 16 * LDIM) * 2;

  // ---- pre-transpose + f16-convert weights into contiguous per-layer blobs
  for (int i = 0; i < NLAYERS; ++i) {
    k_prep<<<(128 * EINP  + 255) / 256, 256, 0, stream>>>(
        msg_w1 + (size_t)i * EIN * LDIM,  wedge + (size_t)i * EDGE_BLOB_H, EIN, EINP);
    k_prep<<<(128 * 128   + 255) / 256, 256, 0, stream>>>(
        msg_w2 + (size_t)i * LDIM * LDIM, wedge + (size_t)i * EDGE_BLOB_H + 128 * EINP, 128, 128);
    k_prep<<<(128 * KNODE + 255) / 256, 256, 0, stream>>>(
        node_w1 + (size_t)i * KNODE * LDIM, wnode + (size_t)i * NODE_BLOB_H, KNODE, KNODE);
    k_prep<<<(128 * 128   + 255) / 256, 256, 0, stream>>>(
        node_w2 + (size_t)i * LDIM * LDIM, wnode + (size_t)i * NODE_BLOB_H + 128 * KNODE, 128, 128);
  }

  // ---- init: attr embed, y0 = x @ W^T, y_old = y0, f16 shadow of y ----
  k_init<<<(NN * LDIM + 255) / 256, 256, 0, stream>>>(
      x_in, node_attr, embed, W, yA, yB, attrh);
  k_half<<<(NN * LDIM / 4 + 255) / 256, 256, 0, stream>>>(yA, yhb);

  float* ycur  = yA;
  float* yprev = yB;
  float* ynext = yC;
  const float* xcur = x_in;

  for (int i = 0; i < NLAYERS; ++i) {
    k_zero<<<(NN * LDIM + 255) / 256, 256, 0, stream>>>(agg, NN * LDIM);

    k_edge<<<320, 256, EDGE_LDS, stream>>>(
        yhb, xcur, attrh, esrc, edst,
        wedge + (size_t)i * EDGE_BLOB_H,
        msg_b1 + (size_t)i * LDIM, msg_b2 + (size_t)i * LDIM,
        agg, EE / 16);

    k_node<<<80, 256, NODE_LDS, stream>>>(
        yhb, ycur, yprev, agg,
        wnode + (size_t)i * NODE_BLOB_H,
        node_b1 + (size_t)i * LDIM, node_b2 + (size_t)i * LDIM,
        hvec, i, ynext, NN / 16);

    // rotate: y_old <- y, y <- y_next
    float* t = yprev;
    yprev = ycur;
    ycur  = ynext;
    ynext = t;

    k_half<<<(NN * LDIM / 4 + 255) / 256, 256, 0, stream>>>(ycur, yhb);
    k_xupd<<<(NN * DIM_IN + 255) / 256, 256, 0, stream>>>(ycur, W, xb);
    xcur = xb;
  }

  k_epi<<<(NN * DIM_IN + 255) / 256, 256, 0, stream>>>(xb, (float*)d_out);
}
